// Router_52415780880435
// MI455X (gfx1250) — matrix operand order repaired and checked by reading, not compiled, not run on hardware
//
#include <hip/hip_runtime.h>
#include <hip/hip_bf16.h>

typedef __attribute__((ext_vector_type(16))) __bf16 v16bf;
typedef __attribute__((ext_vector_type(8)))  __bf16 v8bf;
typedef __attribute__((ext_vector_type(8)))  float  v8f;
typedef __attribute__((ext_vector_type(4)))  float  f32x4;

#define D_DIM 2048
#define E_DIM 64
#define TOK_TOTAL (4 * 8192)
#define TOK_PER_BLOCK 128
#define K_CHUNK 128                 // W staging chunk (fp32 rows per stage)
#define WSTRIDE (K_CHUNK + 8)       // bf16 elems per LDS W row: 272B, 16B-aligned, conflict-free
#define EPAD 65                     // fp32 elems per LDS logit row

#define W_BYTES   (2 * E_DIM * WSTRIDE * 2)       // hi + lo bf16 planes
#define LOG_BYTES (8 * 16 * EPAD * 4)
#define SMEM_BYTES ((W_BYTES) > (LOG_BYTES) ? (W_BYTES) : (LOG_BYTES))

// ---- compare-exchange helpers (descending / ascending), constant-index only
__device__ __forceinline__ void cas_d(unsigned& x, unsigned& y) {
    unsigned mx = x > y ? x : y, mn = x > y ? y : x; x = mx; y = mn;
}
__device__ __forceinline__ void cas_a(unsigned& x, unsigned& y) {
    unsigned mx = x > y ? x : y, mn = x > y ? y : x; x = mn; y = mx;
}
// bitonic sort 8, descending (24 CAS)
__device__ __forceinline__ void sort8_desc(unsigned (&a)[8]) {
    cas_d(a[0],a[1]); cas_a(a[2],a[3]); cas_d(a[4],a[5]); cas_a(a[6],a[7]);
    cas_d(a[0],a[2]); cas_d(a[1],a[3]); cas_a(a[4],a[6]); cas_a(a[5],a[7]);
    cas_d(a[0],a[1]); cas_d(a[2],a[3]); cas_a(a[4],a[5]); cas_a(a[6],a[7]);
    cas_d(a[0],a[4]); cas_d(a[1],a[5]); cas_d(a[2],a[6]); cas_d(a[3],a[7]);
    cas_d(a[0],a[2]); cas_d(a[1],a[3]); cas_d(a[4],a[6]); cas_d(a[5],a[7]);
    cas_d(a[0],a[1]); cas_d(a[2],a[3]); cas_d(a[4],a[5]); cas_d(a[6],a[7]);
}
// top-8 (descending) of two descending 8-lists: half-cleaner + 12-CAS bitonic merge
__device__ __forceinline__ void merge_top8(const unsigned (&a)[8], const unsigned (&b)[8],
                                           unsigned (&r)[8]) {
#pragma unroll
    for (int i = 0; i < 8; ++i) { unsigned u = a[i], v = b[7 - i]; r[i] = u > v ? u : v; }
    cas_d(r[0],r[4]); cas_d(r[1],r[5]); cas_d(r[2],r[6]); cas_d(r[3],r[7]);
    cas_d(r[0],r[2]); cas_d(r[1],r[3]); cas_d(r[4],r[6]); cas_d(r[5],r[7]);
    cas_d(r[0],r[1]); cas_d(r[2],r[3]); cas_d(r[4],r[5]); cas_d(r[6],r[7]);
}

__global__ __launch_bounds__(256) void router_wmma_bf16x3(
    const float* __restrict__ x,      // [B*T, D]
    const float* __restrict__ w,      // [D, E]
    float* __restrict__ out_w,        // [B*T, 8]
    int*   __restrict__ out_i)        // [B*T, 8]
{
    __shared__ __align__(16) unsigned char smem[SMEM_BYTES];
    __bf16* lds_whi = (__bf16*)smem;                         // [E_DIM][WSTRIDE]
    __bf16* lds_wlo = lds_whi + E_DIM * WSTRIDE;
    float*  lds_logit = (float*)smem;                        // reused after GEMM (union)

    const int tid    = threadIdx.x;
    const int wave   = tid >> 5;
    const int lane   = tid & 31;
    const int mA     = lane & 15;        // A-matrix row within 16-token tile
    const int khalfA = lane >> 4;        // A-matrix k-half select
    const int tok0   = blockIdx.x * TOK_PER_BLOCK + wave * 16;

    v8f c0 = {}, c1 = {}, c2 = {}, c3 = {};

    const float* xrow = x + (size_t)(tok0 + mA) * D_DIM;

    // W staging roles: thread handles column wn, 32 rows starting at wkg*32
    const int wn  = tid & 63;
    const int wkg = tid >> 6;            // 0..3

    // B-fragment roles
    const int nb = lane & 15;
    const int kh = (lane >> 4) * 8;      // B k-half*8

    for (int kc0 = 0; kc0 < D_DIM; kc0 += K_CHUNK) {
        __syncthreads();  // previous-stage readers done before overwrite

        // ---- stage W[kc0:kc0+128, 0:64] -> LDS bf16 hi/lo, transposed [n][k]
#pragma unroll
        for (int g = 0; g < K_CHUNK / 32; ++g) {       // 4 groups of 8 rows per thread
            const int rk0 = wkg * 32 + g * 8;
            const float* wp = w + (size_t)(kc0 + rk0) * E_DIM + wn;
            v8bf vh, vl;
#pragma unroll
            for (int j = 0; j < 8; ++j) {
                float v  = wp[(size_t)j * E_DIM];      // coalesced across lanes
                __bf16 h = (__bf16)v;
                vh[j] = h;
                vl[j] = (__bf16)(v - (float)h);
            }
            *(v8bf*)&lds_whi[wn * WSTRIDE + rk0] = vh;
            *(v8bf*)&lds_wlo[wn * WSTRIDE + rk0] = vl;
        }
        __syncthreads();

        // ---- 4 barrier-free K sub-chunks of 32: 48 WMMAs per stage
#pragma unroll
        for (int kcl = 0; kcl < K_CHUNK; kcl += 32) {
            // A fragment: 16-bit A 16x32 layout (streamed once -> non-temporal)
            const int kb = kc0 + kcl + khalfA * 8;
            const f32x4 a0 = __builtin_nontemporal_load((const f32x4*)(xrow + kb));
            const f32x4 a1 = __builtin_nontemporal_load((const f32x4*)(xrow + kb + 4));
            const f32x4 a2 = __builtin_nontemporal_load((const f32x4*)(xrow + kb + 16));
            const f32x4 a3 = __builtin_nontemporal_load((const f32x4*)(xrow + kb + 20));
            float af[16] = {a0[0], a0[1], a0[2], a0[3], a1[0], a1[1], a1[2], a1[3],
                            a2[0], a2[1], a2[2], a2[3], a3[0], a3[1], a3[2], a3[3]};
            v16bf ahi, alo;
#pragma unroll
            for (int i = 0; i < 16; ++i) {
                __bf16 h = (__bf16)af[i];
                ahi[i] = h;
                alo[i] = (__bf16)(af[i] - (float)h);
            }

#pragma unroll
            for (int t = 0; t < 4; ++t) {
                const __bf16* ph = &lds_whi[(t * 16 + nb) * WSTRIDE + kcl + kh];
                const __bf16* pl = &lds_wlo[(t * 16 + nb) * WSTRIDE + kcl + kh];
                v8bf h0 = *(const v8bf*)ph;
                v8bf h1 = *(const v8bf*)(ph + 16);
                v8bf l0 = *(const v8bf*)pl;
                v8bf l1 = *(const v8bf*)(pl + 16);
                v16bf bh, bl;
#pragma unroll
                for (int i = 0; i < 8; ++i) {
                    bh[i] = h0[i]; bh[i + 8] = h1[i];
                    bl[i] = l0[i]; bl[i + 8] = l1[i];
                }
                v8f cc = (t == 0) ? c0 : (t == 1) ? c1 : (t == 2) ? c2 : c3;
                cc = __builtin_amdgcn_wmma_f32_16x16x32_bf16(false, ahi, false, bh,
                                                             (short)0, cc, false, false);
                cc = __builtin_amdgcn_wmma_f32_16x16x32_bf16(false, ahi, false, bl,
                                                             (short)0, cc, false, false);
                cc = __builtin_amdgcn_wmma_f32_16x16x32_bf16(false, alo, false, bh,
                                                             (short)0, cc, false, false);
                if (t == 0) c0 = cc; else if (t == 1) c1 = cc;
                else if (t == 2) c2 = cc; else c3 = cc;
            }
        }
    }

    __syncthreads();  // all W reads done before logit buffer overwrites (union)

    // ---- dump logits: C layout is VGPR r -> M = (lane>=16)*8 + r, lane%16 -> N
    {
        float* lbase   = &lds_logit[wave * 16 * EPAD];
        const int col  = lane & 15;
        const int mofs = (lane >> 4) * 8;
#pragma unroll
        for (int r = 0; r < 8; ++r) {
            lbase[(mofs + r) * EPAD + col]      = c0[r];
            lbase[(mofs + r) * EPAD + col + 16] = c1[r];
            lbase[(mofs + r) * EPAD + col + 32] = c2[r];
            lbase[(mofs + r) * EPAD + col + 48] = c3[r];
        }
    }
    __syncthreads();

    // ---- softmax(64) + top-8 + renorm softmax(8); 2 lanes per token
    {
        const int h     = lane >> 4;             // expert half (0: e 0..31, 1: e 32..63)
        const int ebase = h * 32;
        const float* lr = &lds_logit[wave * 16 * EPAD + mA * EPAD + ebase];

        float pl[32];
        float mxl = -3.4e38f;
#pragma unroll
        for (int j = 0; j < 32; ++j) { pl[j] = lr[j]; mxl = fmaxf(mxl, pl[j]); }
        const float mx = fmaxf(mxl, __shfl_xor(mxl, 16));
        float sl = 0.f;
#pragma unroll
        for (int j = 0; j < 32; ++j) { pl[j] = __expf(pl[j] - mx); sl += pl[j]; }
        const float s   = sl + __shfl_xor(sl, 16);
        const float inv = 1.f / s;

        // pack positive float bits with inverted global index in low 6 bits:
        // unsigned order == (prob desc, then lower expert first) -- jax tie-break
        unsigned g0[8], g1[8], g2[8], g3[8];
#pragma unroll
        for (int j = 0; j < 8; ++j) {
            g0[j] = (__float_as_uint(pl[j])      & ~63u) | (unsigned)(63 - (ebase + j));
            g1[j] = (__float_as_uint(pl[j + 8])  & ~63u) | (unsigned)(63 - (ebase + j + 8));
            g2[j] = (__float_as_uint(pl[j + 16]) & ~63u) | (unsigned)(63 - (ebase + j + 16));
            g3[j] = (__float_as_uint(pl[j + 24]) & ~63u) | (unsigned)(63 - (ebase + j + 24));
        }
        sort8_desc(g0); sort8_desc(g1); sort8_desc(g2); sort8_desc(g3);
        unsigned m0[8], m1[8], loc[8];
        merge_top8(g0, g1, m0);
        merge_top8(g2, g3, m1);
        merge_top8(m0, m1, loc);                 // this half's top-8, descending

        unsigned oth[8];
#pragma unroll
        for (int r = 0; r < 8; ++r) oth[r] = __shfl_xor(loc[r], 16);
        unsigned fin[8];
        merge_top8(loc, oth, fin);               // token's global top-8, descending

        if (lane < 16) {
            const int tok = tok0 + mA;
            float wsel[8];
            int   isel[8];
#pragma unroll
            for (int r = 0; r < 8; ++r) {
                isel[r] = 63 - (int)(fin[r] & 63u);
                wsel[r] = __uint_as_float(fin[r] & ~63u) * inv;  // normalized gating prob
            }
            // softmax over the 8 selected weights (wsel[0] is the max)
            float e8[8], s2 = 0.f;
#pragma unroll
            for (int r = 0; r < 8; ++r) { e8[r] = __expf(wsel[r] - wsel[0]); s2 += e8[r]; }
            const float inv2 = 1.f / s2;
#pragma unroll
            for (int r = 0; r < 8; ++r) {
                out_w[(size_t)tok * 8 + r] = e8[r] * inv2;
                out_i[(size_t)tok * 8 + r] = isel[r];
            }
        }
    }
}

extern "C" void kernel_launch(void* const* d_in, const int* in_sizes, int n_in,
                              void* d_out, int out_size, void* d_ws, size_t ws_size,
                              hipStream_t stream) {
    (void)in_sizes; (void)n_in; (void)d_ws; (void)ws_size; (void)out_size;
    const float* x = (const float*)d_in[0];
    const float* w = (const float*)d_in[1];
    // d_in[2] = num_experts_per_tok (constant 8, baked into the kernel)
    float* out_w = (float*)d_out;                              // [B*T, 8] fp32 weights
    int*   out_i = (int*)d_out + (size_t)TOK_TOTAL * 8;        // [B*T, 8] int32 experts

    router_wmma_bf16x3<<<TOK_TOTAL / TOK_PER_BLOCK, 256, 0, stream>>>(x, w, out_w, out_i);
}
